// MutiHeadSelfAttention_63024350102079
// MI455X (gfx1250) — compile-verified, matrix operand order checked by reading, not run on hardware
//
#include <hip/hip_runtime.h>

// ---------------------------------------------------------------------------
// MHSA for MI455X (gfx1250, wave32, WMMA bf16 16x16x32, f32 accumulate)
// B=4 S=2048 D=1024 H=16 E=64 M=1024
// ---------------------------------------------------------------------------

#define BB 4
#define SS 2048
#define DD 1024
#define HH 16
#define EE 64
#define MM 1024

// gfx1250 async global->LDS path (probe via __has_builtin; safe fallback)
#if defined(__has_builtin)
# if __has_builtin(__builtin_amdgcn_global_load_async_to_lds_b128) && \
     __has_builtin(__builtin_amdgcn_s_wait_asynccnt)
#  define USE_ASYNC_LDS 1
# endif
#endif
#ifndef USE_ASYNC_LDS
# define USE_ASYNC_LDS 0
#endif

typedef __attribute__((ext_vector_type(16))) __bf16 v16bf;
typedef __attribute__((ext_vector_type(8)))  float  v8f;
typedef __attribute__((ext_vector_type(4)))  int    v4i;
typedef unsigned short b16;
typedef v4i as1_v4i __attribute__((address_space(1)));
typedef v4i as3_v4i __attribute__((address_space(3)));

__device__ __forceinline__ b16 f2bf(float f) {
    unsigned u = __builtin_bit_cast(unsigned, f);
    u += 0x7fffu + ((u >> 16) & 1u);          // round-to-nearest-even
    return (b16)(u >> 16);
}

struct FragBF { union { v16bf v; uint4 u[2]; }; };

// Load a K=32 bf16 fragment for one lane. ISA 16-bit A/B layout:
// element i -> k = (i>>3)*16 + 8*(lane>>4) + (i&7)  => two contiguous 16B loads.
__device__ __forceinline__ void ld_frag(FragBF& f, const b16* p, int half) {
    f.u[0] = *reinterpret_cast<const uint4*>(p + 8 * half);
    f.u[1] = *reinterpret_cast<const uint4*>(p + 16 + 8 * half);
}

__device__ __forceinline__ v8f wmma_bf16(const FragBF& a, const FragBF& b, v8f c) {
    return __builtin_amdgcn_wmma_f32_16x16x32_bf16(false, a.v, false, b.v,
                                                   (short)0, c, false, false);
}

// 16-byte global -> LDS copy (per lane); async DMA on gfx1250.
__device__ __forceinline__ void copy16(const b16* g, b16* l) {
#if USE_ASYNC_LDS
    __builtin_amdgcn_global_load_async_to_lds_b128(
        (as1_v4i*)const_cast<b16*>(g), (as3_v4i*)l, 0, 0);
#else
    *reinterpret_cast<uint4*>(l) = *reinterpret_cast<const uint4*>(g);
#endif
}
__device__ __forceinline__ void wait_async() {
#if USE_ASYNC_LDS
    __builtin_amdgcn_s_wait_asynccnt(0);
#endif
}

// ---------------------------------------------------------------------------
// Conversion kernels
// ---------------------------------------------------------------------------
__global__ void cvt_f32_bf16(const float* __restrict__ src, b16* __restrict__ dst, int n) {
    int i = blockIdx.x * 256 + threadIdx.x;
    if (i < n) dst[i] = f2bf(src[i]);
}

// W[D,M] (row-major, d-major) -> WT[M,D] bf16 so B fragments are contiguous in K
__global__ void cvt_transpose(const float* __restrict__ W, b16* __restrict__ WT,
                              int D_, int M_) {
    int i = blockIdx.x * 256 + threadIdx.x;
    if (i < D_ * M_) {
        int d = i / M_;
        int m = i - d * M_;
        WT[(size_t)m * D_ + d] = f2bf(W[i]);
    }
}

// ---------------------------------------------------------------------------
// Fused Q/K/V/R projection GEMM: y = relu(x@W + b) (q,k,v) or x@W + b (r)
// grid = (BS/128, M/128, 4), block = 256 (8 waves); wave -> 32x64 tile (2x4 WMMA)
// ---------------------------------------------------------------------------
__global__ __launch_bounds__(256) void proj_kernel(
    const b16* __restrict__ xb, const b16* __restrict__ wt,
    const float* __restrict__ bq, const float* __restrict__ bk,
    const float* __restrict__ bv, const float* __restrict__ br,
    b16* __restrict__ qb, b16* __restrict__ kb, b16* __restrict__ vt,
    float* __restrict__ outr)
{
    const int lane = threadIdx.x & 31;
    const int wave = threadIdx.x >> 5;
    const int half = lane >> 4;
    const int ln16 = lane & 15;
    const int z    = blockIdx.z;                      // 0=q 1=k 2=v 3=r
    const int rbase = blockIdx.x * 128 + (wave >> 1) * 32;
    const int cbase = blockIdx.y * 128 + (wave & 1) * 64;

    const b16*   w    = wt + (size_t)z * ((size_t)DD * MM);
    const float* bias = (z == 0) ? bq : (z == 1) ? bk : (z == 2) ? bv : br;

    v8f acc[2][4];
    for (int i = 0; i < 2; ++i)
        for (int j = 0; j < 4; ++j) acc[i][j] = (v8f){};

    const b16* arow0 = xb + (size_t)(rbase + ln16) * DD;
    const b16* arow1 = arow0 + (size_t)16 * DD;
    const b16* bcol  = w + (size_t)(cbase + ln16) * DD;

    for (int k = 0; k < DD; k += 32) {
        FragBF a0, a1, bf[4];
        ld_frag(a0, arow0 + k, half);
        ld_frag(a1, arow1 + k, half);
        for (int j = 0; j < 4; ++j)
            ld_frag(bf[j], bcol + (size_t)j * 16 * DD + k, half);
        for (int j = 0; j < 4; ++j) {
            acc[0][j] = wmma_bf16(a0, bf[j], acc[0][j]);
            acc[1][j] = wmma_bf16(a1, bf[j], acc[1][j]);
        }
    }

    for (int ti = 0; ti < 2; ++ti)
        for (int tj = 0; tj < 4; ++tj) {
            int coln = cbase + tj * 16 + ln16;        // C layout: lane -> N
            float bv_ = bias[coln];
            for (int j = 0; j < 8; ++j) {            // VGPR j -> M = j + 8*half
                int row = rbase + ti * 16 + j + 8 * half;
                float val = acc[ti][tj][j] + bv_;
                if (z < 3) val = fmaxf(val, 0.0f);
                size_t idx = (size_t)row * MM + coln;
                if (z == 0)      qb[idx] = f2bf(val);
                else if (z == 1) kb[idx] = f2bf(val);
                else if (z == 2) {                   // V transposed: [B,H,E,S]
                    int bi = row >> 11, s = row & (SS - 1);
                    int h = coln >> 6, e = coln & 63;
                    vt[(((size_t)bi * HH + h) * EE + e) * SS + s] = f2bf(val);
                } else outr[idx] = val;              // residual written to d_out
            }
        }
}

// ---------------------------------------------------------------------------
// Flash attention: 4 waves per block, 64 q-rows; K/V tiles staged in LDS once
// per block via async global->LDS DMA. Online softmax, fp32 stats.
// out[b,s,h*E+e] = residual(read from d_out) + softmax(q kT) v
// ---------------------------------------------------------------------------
__global__ __launch_bounds__(128) void attn_kernel(
    const b16* __restrict__ qb, const b16* __restrict__ kb,
    const b16* __restrict__ vt, float* __restrict__ out)
{
    __shared__ __align__(16) b16 kT[32][64];        // [k-row][e]
    __shared__ __align__(16) b16 vS[64][32];        // [e][k-row]  (from v^T)
    __shared__ __align__(16) b16 sP[4][16][32];     // per-wave P bounce

    const int tid  = threadIdx.x;
    const int lane = tid & 31;
    const int wave = tid >> 5;
    const int half = lane >> 4;
    const int ln16 = lane & 15;
    const int qbase = blockIdx.x * 64 + wave * 16;
    const int h = blockIdx.y;
    const int b = blockIdx.z;
    const size_t rowBase = (size_t)b * SS;

    FragBF qf0, qf1;                                // q tile 16x64: K=64 in 2 frags
    {
        const b16* qrow = qb + (rowBase + qbase + ln16) * MM + h * EE;
        ld_frag(qf0, qrow,      half);
        ld_frag(qf1, qrow + 32, half);
    }
    const b16* khead = kb + rowBase * MM + h * EE;
    const b16* vhead = vt + ((size_t)(b * HH + h) * EE) * SS;

    v8f o0 = {}, o1 = {}, o2 = {}, o3 = {};
    float mi[8], li[8];
    for (int j = 0; j < 8; ++j) { mi[j] = -1e30f; li[j] = 0.0f; }

    for (int kk = 0; kk < SS; kk += 32) {
        // ---- cooperative stage: K tile 32x64, V tile 64x32 (8 KB) ---------
        for (int it = 0; it < 2; ++it) {
            int chunk = it * 128 + tid;             // 256 x 16B for kT
            int r = chunk >> 3, c = (chunk & 7) * 8;
            copy16(khead + (size_t)(kk + r) * MM + c, &kT[r][c]);
        }
        for (int it = 0; it < 2; ++it) {
            int chunk = it * 128 + tid;             // 256 x 16B for vS
            int e = chunk >> 2, c = (chunk & 3) * 8;
            copy16(vhead + (size_t)e * SS + kk + c, &vS[e][c]);
        }
        wait_async();
        __syncthreads();

        // ---- scores s[16q][32k]: B = k^T, col n = k-row (from LDS) --------
        FragBF kf;
        v8f s0 = {}, s1 = {};
        ld_frag(kf, &kT[ln16][0],       half); s0 = wmma_bf16(qf0, kf, s0);
        ld_frag(kf, &kT[ln16][32],      half); s0 = wmma_bf16(qf1, kf, s0);
        ld_frag(kf, &kT[16 + ln16][0],  half); s1 = wmma_bf16(qf0, kf, s1);
        ld_frag(kf, &kT[16 + ln16][32], half); s1 = wmma_bf16(qf1, kf, s1);

        // ---- online softmax; row m=j+8*half lives in this 16-lane half ----
        float p0[8], p1[8], sc[8];
        for (int j = 0; j < 8; ++j) {
            float t = fmaxf(s0[j], s1[j]);
            for (int off = 1; off < 16; off <<= 1) t = fmaxf(t, __shfl_xor(t, off, 32));
            float mnew = fmaxf(mi[j], t);
            p0[j] = __expf(s0[j] - mnew);
            p1[j] = __expf(s1[j] - mnew);
            float rs = p0[j] + p1[j];
            for (int off = 1; off < 16; off <<= 1) rs += __shfl_xor(rs, off, 32);
            sc[j] = __expf(mi[j] - mnew);
            li[j] = li[j] * sc[j] + rs;
            mi[j] = mnew;
        }

        // ---- re-layout P (C-layout f32) -> A-layout bf16 via LDS ----------
        for (int j = 0; j < 8; ++j) {
            sP[wave][j + 8 * half][ln16]      = f2bf(p0[j]);
            sP[wave][j + 8 * half][16 + ln16] = f2bf(p1[j]);
        }
        FragBF pf;
        pf.u[0] = *reinterpret_cast<const uint4*>(&sP[wave][ln16][8 * half]);
        pf.u[1] = *reinterpret_cast<const uint4*>(&sP[wave][ln16][16 + 8 * half]);

        for (int j = 0; j < 8; ++j) {
            o0[j] *= sc[j]; o1[j] *= sc[j]; o2[j] *= sc[j]; o3[j] *= sc[j];
        }

        // ---- O += P(16x32) @ V(32x64), V fragments from LDS ---------------
        FragBF vf;
        ld_frag(vf, &vS[0 * 16 + ln16][0], half); o0 = wmma_bf16(pf, vf, o0);
        ld_frag(vf, &vS[1 * 16 + ln16][0], half); o1 = wmma_bf16(pf, vf, o1);
        ld_frag(vf, &vS[2 * 16 + ln16][0], half); o2 = wmma_bf16(pf, vf, o2);
        ld_frag(vf, &vS[3 * 16 + ln16][0], half); o3 = wmma_bf16(pf, vf, o3);
        __syncthreads();                            // before next stage overwrites
    }

    // ---- normalize and add the residual already sitting in d_out ----------
    for (int j = 0; j < 8; ++j) {
        float inv = 1.0f / li[j];
        size_t base = (rowBase + qbase + j + 8 * half) * MM + (size_t)h * EE;
        size_t i0 = base + 0 * 16 + ln16; out[i0] += o0[j] * inv;
        size_t i1 = base + 1 * 16 + ln16; out[i1] += o1[j] * inv;
        size_t i2 = base + 2 * 16 + ln16; out[i2] += o2[j] * inv;
        size_t i3 = base + 3 * 16 + ln16; out[i3] += o3[j] * inv;
    }
}

// ---------------------------------------------------------------------------
extern "C" void kernel_launch(void* const* d_in, const int* in_sizes, int n_in,
                              void* d_out, int out_size, void* d_ws, size_t ws_size,
                              hipStream_t stream)
{
    (void)in_sizes; (void)n_in; (void)out_size; (void)ws_size;
    const float* x  = (const float*)d_in[0];
    const float* Wq = (const float*)d_in[1];
    const float* bq = (const float*)d_in[2];
    const float* Wk = (const float*)d_in[3];
    const float* bk = (const float*)d_in[4];
    const float* Wv = (const float*)d_in[5];
    const float* bv = (const float*)d_in[6];
    const float* Wr = (const float*)d_in[7];
    const float* br = (const float*)d_in[8];
    float* out = (float*)d_out;

    // workspace layout (bf16 elements): x | 4xW^T | q | k | v^T   (~75 MB)
    b16* xb   = (b16*)d_ws;
    b16* wt   = xb   + (size_t)BB * SS * DD;
    b16* qb   = wt   + (size_t)4 * DD * MM;
    b16* kbuf = qb   + (size_t)BB * SS * MM;
    b16* vtb  = kbuf + (size_t)BB * SS * MM;

    const int nX = BB * SS * DD;
    cvt_f32_bf16<<<(nX + 255) / 256, 256, 0, stream>>>(x, xb, nX);
    const int nW = DD * MM;
    cvt_transpose<<<(nW + 255) / 256, 256, 0, stream>>>(Wq, wt + 0 * (size_t)nW, DD, MM);
    cvt_transpose<<<(nW + 255) / 256, 256, 0, stream>>>(Wk, wt + 1 * (size_t)nW, DD, MM);
    cvt_transpose<<<(nW + 255) / 256, 256, 0, stream>>>(Wv, wt + 2 * (size_t)nW, DD, MM);
    cvt_transpose<<<(nW + 255) / 256, 256, 0, stream>>>(Wr, wt + 3 * (size_t)nW, DD, MM);

    dim3 pg(BB * SS / 128, MM / 128, 4);
    proj_kernel<<<pg, 256, 0, stream>>>(xb, wt, bq, bk, bv, br, qb, kbuf, vtb, out);

    dim3 ag(SS / 64, HH, BB);
    attn_kernel<<<ag, 128, 0, stream>>>(qb, kbuf, vtb, out);
}
